// ScannedRNN_51677046505596
// MI455X (gfx1250) — matrix operand order repaired and checked by reading, not compiled, not run on hardware
//
#include <hip/hip_runtime.h>
#include <hip/hip_bf16.h>
#include <math.h>

// ============================================================================
// 2-layer LayerNorm-GRU scan for MI455X (gfx1250).
//
// Strategy: the scan is batch-row independent, so 8 persistent workgroups each
// own 16 batch rows for all 512 timesteps (no inter-WG sync). All GEMMs run as
// v_wmma_f32_16x16x32_bf16 (bf16 A/B, f32 accum). Weights are converted once
// to bf16, stored N-major (transposed) in d_ws so each lane's B-operand load
// is two contiguous 16-byte runs; cell-1's input kernel Wi[1][:, :1024] is pre-summed
// with Wh_rz[1] since both multiply h1. The f32 hidden state lives in VGPRs
// across the whole scan; a bf16 shadow is staged to LDS per step as WMMA A.
// ============================================================================

typedef __bf16 bf16t;
typedef __attribute__((ext_vector_type(16))) __bf16 v16bf;
typedef __attribute__((ext_vector_type(8)))  __bf16 v8bf;
typedef __attribute__((ext_vector_type(8)))  float  v8f;

#define TT 512
#define BB 128
#define HH 512
#define LN_EPS 1e-6f

// bf16 workspace layout (element offsets):
//   w0x : [1536][512]  = transpose(Wi[0])                  (A = x)
//   w0h : [1536][512]  = transpose([Wh_rz[0] | Wh_n[0]])   (A = h)
//   w1  : [2048][512]  = transpose([Wi1_rz+Wh_rz1 | Wi1_n | Wh_n1]) (A = h1)
#define W0X_OFF 0
#define W0H_OFF (1536 * 512)
#define W1_OFF  (2 * 1536 * 512)
#define WS_ELEMS (W1_OFF + 2048 * 512)

// ---------------------------------------------------------------------------
// Weight prep: fp32 -> bf16, transpose to N-major, fuse cell-1 rz kernels.
// ---------------------------------------------------------------------------
__global__ void prep_weights(const float* __restrict__ Wi,     // [2][512][1536]
                             const float* __restrict__ Wh_rz,  // [2][512][1024]
                             const float* __restrict__ Wh_n,   // [2][512][512]
                             bf16t* __restrict__ ws) {
  int idx = blockIdx.x * blockDim.x + threadIdx.x;
  const int total = 5120 * 512;
  if (idx >= total) return;
  int k = idx & 511;
  int r = idx >> 9;  // output row (N index across the three packed matrices)
  float v;
  bf16t* dst;
  if (r < 1536) {                                   // w0x
    v = Wi[k * 1536 + r];
    dst = ws + W0X_OFF + r * 512 + k;
  } else if (r < 3072) {                            // w0h
    int n = r - 1536;
    v = (n < 1024) ? Wh_rz[k * 1024 + n] : Wh_n[k * 512 + (n - 1024)];
    dst = ws + W0H_OFF + n * 512 + k;
  } else {                                          // w1 (cell 1, fused)
    int n = r - 3072;
    if (n < 1024)
      v = Wi[512 * 1536 + k * 1536 + n] + Wh_rz[512 * 1024 + k * 1024 + n];
    else if (n < 1536)
      v = Wi[512 * 1536 + k * 1536 + 1024 + (n - 1024)];
    else
      v = Wh_n[512 * 512 + k * 512 + (n - 1536)];
    dst = ws + W1_OFF + n * 512 + k;
  }
  *dst = (bf16t)v;
}

// ---------------------------------------------------------------------------
// WMMA operand loaders (bf16 16x16x32, wave32 layouts from CDNA5 ISA)
// A (16x32, from LDS): lane<16 -> row=lane, K {k0..k0+7, k0+16..k0+23}
//                      lane>=16 -> row=lane-16, K {k0+8..15, k0+24..31}
// B (32x16, from N-major global): lane<16 -> col=lane, K {k0..k0+7, k0+16..k0+23}
//                      lane>=16 -> col=lane-16, K {k0+8..15, k0+24..31}
// ---------------------------------------------------------------------------
__device__ __forceinline__ v16bf load_a(const bf16t* p0, int lane, int k0) {
  const bf16t* p = p0 + (lane & 15) * HH + k0 + ((lane >> 4) << 3);
  v8bf lo = *(const v8bf*)(p);        // ds_load_b128
  v8bf hi = *(const v8bf*)(p + 16);   // ds_load_b128
  v16bf a;
#pragma unroll
  for (int i = 0; i < 8; ++i) { a[i] = lo[i]; a[8 + i] = hi[i]; }
  return a;
}

__device__ __forceinline__ v16bf load_b(const bf16t* __restrict__ wt, int lane,
                                        int n0, int k0) {
  const bf16t* p = wt + (size_t)(n0 + (lane & 15)) * HH + k0 + ((lane >> 4) << 3);
  v8bf lo = *(const v8bf*)(p);        // global_load_b128 (L2 resident)
  v8bf hi = *(const v8bf*)(p + 16);
  v16bf b;
#pragma unroll
  for (int i = 0; i < 8; ++i) { b[i] = lo[i]; b[8 + i] = hi[i]; }
  return b;
}

__device__ __forceinline__ v8f wmma_bf16(v16bf a, v16bf b, v8f c) {
  return __builtin_amdgcn_wmma_f32_16x16x32_bf16(false, a, false, b,
                                                 (short)0, c, false, false);
}

// Row-sum / row-sumsq of 4 C-tiles into LDS accumulators (for LayerNorm).
// C layout: lane holds col, VGPR v holds row v + 8*(lane>=16). Reduce over the
// 16 columns with a shfl_xor butterfly (stays within each 16-lane half).
__device__ __forceinline__ void ln_accum(const v8f* acc4, float* sumRow,
                                         float* sqRow, int lane) {
  float s[8], q[8];
#pragma unroll
  for (int v = 0; v < 8; ++v) { s[v] = 0.f; q[v] = 0.f; }
#pragma unroll
  for (int j = 0; j < 4; ++j)
#pragma unroll
    for (int v = 0; v < 8; ++v) { float x = acc4[j][v]; s[v] += x; q[v] += x * x; }
#pragma unroll
  for (int m = 1; m < 16; m <<= 1)
#pragma unroll
    for (int v = 0; v < 8; ++v) {
      s[v] += __shfl_xor(s[v], m, 32);
      q[v] += __shfl_xor(q[v], m, 32);
    }
  if ((lane & 15) == 0) {
    int hi = lane >> 4;
#pragma unroll
    for (int v = 0; v < 8; ++v) {
      atomicAdd(&sumRow[v + 8 * hi], s[v]);   // ds_add_f32
      atomicAdd(&sqRow[v + 8 * hi], q[v]);
    }
  }
}

__device__ __forceinline__ float sigmoidf_(float x) {
  return 1.0f / (1.0f + __expf(-x));
}

// ---------------------------------------------------------------------------
// Persistent scan kernel: gridDim.x = 8, blockDim.x = 256 (8 wave32 waves).
// Block b owns batch rows [16b, 16b+16); wave w owns gate columns [64w, 64w+64).
// ---------------------------------------------------------------------------
__global__ void __launch_bounds__(256)
gru_scan(const float* __restrict__ ins,          // [T][B][H]
         const unsigned char* __restrict__ resets,// [T][B] (bool)
         const float* __restrict__ h0,           // [B][H]
         const float* __restrict__ bi,           // [2][1536]
         const float* __restrict__ bh_n,         // [2][512]
         const float* __restrict__ ln_scale,     // [2][2][512]
         const float* __restrict__ ln_bias,      // [2][2][512]
         const bf16t* __restrict__ ws,
         float* __restrict__ out)                // [B*H (h_final)] ++ [T*B*H (ys)]
{
  __shared__ bf16t s_x[16 * HH];    // x_t tile, bf16
  __shared__ bf16t s_h[16 * HH];    // bf16 shadow of carried h
  __shared__ bf16t s_h1[16 * HH];   // bf16 shadow of cell-0 output
  __shared__ float s_sum[2][2][16]; // [cell][gate r/z][row]
  __shared__ float s_sq[2][2][16];

  const int tid  = threadIdx.x;
  const int lane = tid & 31;
  const int wave = tid >> 5;
  const int b0   = blockIdx.x * 16;
  const int c0   = wave * 64;
  const int col  = c0 + (lane & 15);  // column for tile j=0
  const int hi   = (lane >> 4) & 1;

  const bf16t* w0x = ws + W0X_OFF;
  const bf16t* w0h = ws + W0H_OFF;
  const bf16t* w1  = ws + W1_OFF;

  // Hoist per-column constants (fixed for the whole scan).
  float b_r[4], b_z[4], b_xn[4], b_hn[4];
  float b_r2[4], b_z2[4], b_xn2[4], b_hn2[4];
  float lns[2][2][4], lnb[2][2][4];
#pragma unroll
  for (int j = 0; j < 4; ++j) {
    int n = col + 16 * j;
    b_r[j]   = bi[n];
    b_z[j]   = bi[512 + n];
    b_xn[j]  = bi[1024 + n];
    b_hn[j]  = bh_n[n];
    b_r2[j]  = bi[1536 + n];
    b_z2[j]  = bi[1536 + 512 + n];
    b_xn2[j] = bi[1536 + 1024 + n];
    b_hn2[j] = bh_n[512 + n];
    for (int c = 0; c < 2; ++c)
      for (int g = 0; g < 2; ++g) {
        lns[c][g][j] = ln_scale[(c * 2 + g) * 512 + n];
        lnb[c][g][j] = ln_bias[(c * 2 + g) * 512 + n];
      }
  }

  // f32 hidden state carried in VGPRs for all 512 steps (C-matrix layout).
  v8f h[4];
#pragma unroll
  for (int j = 0; j < 4; ++j)
#pragma unroll
    for (int v = 0; v < 8; ++v)
      h[j][v] = h0[(size_t)(b0 + v + 8 * hi) * HH + col + 16 * j];

  const float inv = 1.0f / (float)HH;

  for (int t = 0; t < TT; ++t) {
    // ---- phase A: reset carry, stage bf16 h and x_t into LDS --------------
    if (tid < 64) {
      ((float*)s_sum)[tid] = 0.f;
      ((float*)s_sq)[tid]  = 0.f;
    }
#pragma unroll
    for (int v = 0; v < 8; ++v) {
      int row = v + 8 * hi;
      if (resets[(size_t)t * BB + b0 + row]) {
#pragma unroll
        for (int j = 0; j < 4; ++j) h[j][v] = 0.f;
      }
    }
#pragma unroll
    for (int j = 0; j < 4; ++j)
#pragma unroll
      for (int v = 0; v < 8; ++v)
        s_h[(v + 8 * hi) * HH + col + 16 * j] = (bf16t)h[j][v];

    {
      const float* src = ins + ((size_t)t * BB + b0) * HH;
      for (int i = tid; i < 16 * HH; i += 256) s_x[i] = (bf16t)src[i];
    }
    if (t + 1 < TT)  // hint next timestep's input tile toward L2
      __builtin_prefetch(ins + ((size_t)(t + 1) * BB + b0) * HH, 0, 1);
    __syncthreads();

    // ---- cell 0: pre_r/pre_z = x@Wi0 + h@Wh_rz0 + bias; xn, hn ------------
    v8f ar[4], az[4], axn[4], ahn[4];
#pragma unroll
    for (int j = 0; j < 4; ++j)
#pragma unroll
      for (int v = 0; v < 8; ++v) {
        ar[j][v] = b_r[j];  az[j][v] = b_z[j];
        axn[j][v] = b_xn[j]; ahn[j][v] = b_hn[j];
      }
    for (int k0 = 0; k0 < HH; k0 += 32) {
      v16bf axv = load_a(s_x, lane, k0);
      v16bf ahv = load_a(s_h, lane, k0);
#pragma unroll
      for (int j = 0; j < 4; ++j) {
        int n = c0 + 16 * j;
        ar[j]  = wmma_bf16(axv, load_b(w0x, lane, n, k0),        ar[j]);
        ar[j]  = wmma_bf16(ahv, load_b(w0h, lane, n, k0),        ar[j]);
        az[j]  = wmma_bf16(axv, load_b(w0x, lane, 512 + n, k0),  az[j]);
        az[j]  = wmma_bf16(ahv, load_b(w0h, lane, 512 + n, k0),  az[j]);
        axn[j] = wmma_bf16(axv, load_b(w0x, lane, 1024 + n, k0), axn[j]);
        ahn[j] = wmma_bf16(ahv, load_b(w0h, lane, 1024 + n, k0), ahn[j]);
      }
    }
    ln_accum(ar, s_sum[0][0], s_sq[0][0], lane);
    ln_accum(az, s_sum[0][1], s_sq[0][1], lane);
    __syncthreads();

    // gates + h1 (registers); stage bf16 h1 for cell-1 A operand
#pragma unroll
    for (int j = 0; j < 4; ++j)
#pragma unroll
      for (int v = 0; v < 8; ++v) {
        int row = v + 8 * hi;
        float mr = s_sum[0][0][row] * inv;
        float vr = s_sq[0][0][row] * inv - mr * mr;
        float r  = sigmoidf_((ar[j][v] - mr) * rsqrtf(vr + LN_EPS) * lns[0][0][j] + lnb[0][0][j]);
        float mz = s_sum[0][1][row] * inv;
        float vz = s_sq[0][1][row] * inv - mz * mz;
        float z  = sigmoidf_((az[j][v] - mz) * rsqrtf(vz + LN_EPS) * lns[0][1][j] + lnb[0][1][j]);
        float nn = tanhf(axn[j][v] + r * ahn[j][v]);
        float h1 = (1.0f - z) * nn + z * h[j][v];
        h[j][v] = h1;  // carry slot now holds h1
        s_h1[row * HH + col + 16 * j] = (bf16t)h1;
      }
    __syncthreads();

    // ---- cell 1: A = h1; rz-kernels pre-fused in w1 -----------------------
#pragma unroll
    for (int j = 0; j < 4; ++j)
#pragma unroll
      for (int v = 0; v < 8; ++v) {
        ar[j][v] = b_r2[j];  az[j][v] = b_z2[j];
        axn[j][v] = b_xn2[j]; ahn[j][v] = b_hn2[j];
      }
    for (int k0 = 0; k0 < HH; k0 += 32) {
      v16bf ahv = load_a(s_h1, lane, k0);
#pragma unroll
      for (int j = 0; j < 4; ++j) {
        int n = c0 + 16 * j;
        ar[j]  = wmma_bf16(ahv, load_b(w1, lane, n, k0),         ar[j]);
        az[j]  = wmma_bf16(ahv, load_b(w1, lane, 512 + n, k0),   az[j]);
        axn[j] = wmma_bf16(ahv, load_b(w1, lane, 1024 + n, k0),  axn[j]);
        ahn[j] = wmma_bf16(ahv, load_b(w1, lane, 1536 + n, k0),  ahn[j]);
      }
    }
    ln_accum(ar, s_sum[1][0], s_sq[1][0], lane);
    ln_accum(az, s_sum[1][1], s_sq[1][1], lane);
    __syncthreads();

    // gates + h2; write ys[t] (16 lanes -> 64 contiguous bytes per row)
    float* ys = out + (size_t)BB * HH + (size_t)t * BB * HH;
#pragma unroll
    for (int j = 0; j < 4; ++j)
#pragma unroll
      for (int v = 0; v < 8; ++v) {
        int row = v + 8 * hi;
        float mr = s_sum[1][0][row] * inv;
        float vr = s_sq[1][0][row] * inv - mr * mr;
        float r2 = sigmoidf_((ar[j][v] - mr) * rsqrtf(vr + LN_EPS) * lns[1][0][j] + lnb[1][0][j]);
        float mz = s_sum[1][1][row] * inv;
        float vz = s_sq[1][1][row] * inv - mz * mz;
        float z2 = sigmoidf_((az[j][v] - mz) * rsqrtf(vz + LN_EPS) * lns[1][1][j] + lnb[1][1][j]);
        float n2 = tanhf(axn[j][v] + r2 * ahn[j][v]);
        float h2 = (1.0f - z2) * n2 + z2 * h[j][v];
        h[j][v] = h2;
        ys[(size_t)(b0 + row) * HH + col + 16 * j] = h2;
        if (t == TT - 1)
          out[(size_t)(b0 + row) * HH + col + 16 * j] = h2;  // h_final
      }
    __syncthreads();  // protect LN buffers / LDS tiles before next step
  }
}

// ---------------------------------------------------------------------------
extern "C" void kernel_launch(void* const* d_in, const int* in_sizes, int n_in,
                              void* d_out, int out_size, void* d_ws, size_t ws_size,
                              hipStream_t stream) {
  const float*         ins      = (const float*)d_in[0];
  const unsigned char* resets   = (const unsigned char*)d_in[1];  // bool
  const float*         h0       = (const float*)d_in[2];
  const float*         Wi       = (const float*)d_in[3];
  const float*         bi       = (const float*)d_in[4];
  const float*         Wh_rz    = (const float*)d_in[5];
  const float*         Wh_n     = (const float*)d_in[6];
  const float*         bh_n     = (const float*)d_in[7];
  const float*         ln_scale = (const float*)d_in[8];
  const float*         ln_bias  = (const float*)d_in[9];
  float*               out      = (float*)d_out;
  bf16t*               ws       = (bf16t*)d_ws;   // needs WS_ELEMS*2 = ~5.25 MB

  {
    const int total = 5120 * 512;
    prep_weights<<<(total + 255) / 256, 256, 0, stream>>>(Wi, Wh_rz, Wh_n, ws);
  }
  gru_scan<<<BB / 16, 256, 0, stream>>>(ins, resets, h0, bi, bh_n,
                                        ln_scale, ln_bias, ws, out);
}
